// DVHLoss_48077863912000
// MI455X (gfx1250) — matrix operand order repaired and checked by reading, not compiled, not run on hardware
//
#include <hip/hip_runtime.h>
#include <math.h>

// ---------------------------------------------------------------------------
// DVH loss for MI455X (gfx1250, wave32, WMMA)
//
// out = (1/(85*22)) * sum_s sqrt( sum_d [ (sum_n delta[n,d]*mask[n,s]) / tot[s] ]^2 )
// delta[n,d] = sigmoid(dose_t[n]-d) - sigmoid(relu(dose_p[n])-d)
//            = 1/(1+e^d * e^{-dose_t}) - 1/(1+e^d * e^{-dose_p})
//
// Contraction = (22 x 2M) x (2M x 85) GEMM via v_wmma_f32_16x16x32_f16,
// K-chunks of 32 samples per wave. Pad column d=85 of B is held at 1.0 so
// C[s,85] = sum_n mask[n,s] = tot_vol[s] -- the mask totals come free from
// the same WMMAs (no extra reduction path).
// Sigmoids use raw v_rcp_f32 (result is truncated to f16 anyway), keeping
// the 85-bin loop at ~4 VALU ops/bin so the kernel stays HBM-bound (~16us).
// ---------------------------------------------------------------------------

typedef __attribute__((ext_vector_type(16))) _Float16 v16h;
typedef __attribute__((ext_vector_type(8)))  float    v8f;

#define NSTOT 23      // columns of y_true / y_pred
#define NSM   22      // structures (masks)
#define NDR   85      // real dose bins
#define NDP   96      // padded bins  (6 x 16 WMMA tiles); col 85 = tot_vol
#define NSP   32      // padded structures (2 x 16 WMMA tiles)
#define WPB   4       // waves per block
#define BLOCK (WPB * 32)
#define ACC_ELEMS (NSP * NDP)

union Frag16 { uint4 q[2]; v16h v; };

__global__ void dvh_zero(float* __restrict__ ws) {
  int i = blockIdx.x * blockDim.x + threadIdx.x;
  if (i < ACC_ELEMS) ws[i] = 0.0f;
}

__global__ __launch_bounds__(BLOCK) void dvh_accum(
    const float* __restrict__ y_true,   // [N][23]
    const float* __restrict__ y_pred,   // [N][23]
    float* __restrict__ acc,            // [NSP][NDP]  (workspace, zeroed)
    int N)
{
  // Per-wave private LDS staging slices (no cross-wave sharing -> no s_barrier).
  __shared__ _Float16 s_delta[WPB][NDP * 32];  // [d][n]  transposed deltas
  __shared__ _Float16 s_mask [WPB][NSP * 32];  // [s][n]  transposed masks

  const int lane = threadIdx.x & 31;
  const int wid  = threadIdx.x >> 5;
  _Float16* dT = s_delta[wid];
  _Float16* mT = s_mask[wid];

  // Pad rows written once; real rows rewritten every chunk.
  // Row d=85 is the all-ones column that makes C[s,85] = tot_vol[s].
  dT[NDR * 32 + lane] = (_Float16)1.0f;
  for (int d = NDR + 1; d < NDP; ++d) dT[d * 32 + lane] = (_Float16)0.0f;
  for (int s = NSM; s < NSP; ++s)     mT[s * 32 + lane] = (_Float16)0.0f;

  v8f c[2][6];
#pragma unroll
  for (int st = 0; st < 2; ++st)
#pragma unroll
    for (int t = 0; t < 6; ++t) { v8f z = {}; c[st][t] = z; }

  const int wave    = blockIdx.x * WPB + wid;
  const int wstride = gridDim.x * WPB;
  const int nChunks = (N + 31) >> 5;

  const int M   = lane & 15;             // row/col within 16x16 tile
  const int kbA = (lane < 16) ? 0 : 8;   // A-fragment K base (16-bit A 16x32)
  const int kbB = (lane < 16) ? 0 : 8;   // B-fragment K base (16-bit B 32x16)

  for (int chunk = wave; chunk < nChunks; chunk += wstride) {
    const int n      = (chunk << 5) + lane;
    const bool valid = (n < N);
    const float* yt  = y_true + (size_t)n * NSTOT;

    // Prefetch next chunk's rows (global_prefetch_b8).
    if (chunk + wstride < nChunks) {
      __builtin_prefetch(y_true + ((size_t)((chunk + wstride) << 5) + lane) * NSTOT, 0, 1);
    }

    float dose_t = 0.0f, dose_p = 0.0f;
    if (valid) {
      dose_t = yt[0];
      dose_p = fmaxf(y_pred[(size_t)n * NSTOT], 0.0f);  // relu
    }
    const float et = __expf(-dose_t);
    const float ep = __expf(-dose_p);

    // delta[d] = rcp(1+e^d*et) - rcp(1+e^d*ep); e^d built incrementally.
    // Raw v_rcp_f32 (~1 ulp) is plenty: result is truncated to f16.
    float Ed = 1.0f;
#pragma unroll 5
    for (int d = 0; d < NDR; ++d) {
      float st = __builtin_amdgcn_rcpf(fmaf(Ed, et, 1.0f));
      float sp = __builtin_amdgcn_rcpf(fmaf(Ed, ep, 1.0f));
      dT[d * 32 + lane] = (_Float16)(st - sp);   // zero when !valid (et==ep)
      Ed *= 2.71828182845904523f;
    }

#pragma unroll
    for (int s = 0; s < NSM; ++s) {
      float m = 0.0f;
      if (valid) m = (yt[1 + s] != 0.0f) ? 1.0f : 0.0f;  // 0 when !valid
      mT[s * 32 + lane] = (_Float16)m;
    }

    // DS ops are in-order within a wave; fence the compiler + drain DScnt
    // so fragment reads see this chunk's staged data.
    asm volatile("s_wait_dscnt 0" ::: "memory");

    // A fragments (masks): halves i -> K = (i<8?0:16) + kbA + (i&7)
    //   => two contiguous 16B runs per lane.
    Frag16 a0, a1;
    {
      const _Float16* r0 = mT + (0 * 16 + M) * 32;
      a0.q[0] = *(const uint4*)(r0 + kbA);
      a0.q[1] = *(const uint4*)(r0 + 16 + kbA);
      const _Float16* r1 = mT + (1 * 16 + M) * 32;
      a1.q[0] = *(const uint4*)(r1 + kbA);
      a1.q[1] = *(const uint4*)(r1 + 16 + kbA);
    }

#pragma unroll
    for (int t = 0; t < 6; ++t) {
      // B fragment (deltas): halves i -> K = (i<8?0:16) + kbB + (i&7), two 16B runs.
      Frag16 b;
      const _Float16* bp = dT + (t * 16 + M) * 32 + kbB;
      b.q[0] = *(const uint4*)(bp);
      b.q[1] = *(const uint4*)(bp + 16);

      c[0][t] = __builtin_amdgcn_wmma_f32_16x16x32_f16(
          false, a0.v, false, b.v, (short)0, c[0][t], false, false);
      c[1][t] = __builtin_amdgcn_wmma_f32_16x16x32_f16(
          false, a1.v, false, b.v, (short)0, c[1][t], false, false);
    }

    // Keep next iteration's LDS stores after this iteration's reads.
    asm volatile("" ::: "memory");
  }

  // Flush accumulators. C/D layout: VGPR r, lanes 0-15 -> row M=r, lanes
  // 16-31 -> row M=r+8; column N = lane&15.
  const int rowOff = (lane < 16) ? 0 : 8;
#pragma unroll
  for (int st = 0; st < 2; ++st)
#pragma unroll
    for (int t = 0; t < 6; ++t)
#pragma unroll
      for (int r = 0; r < 8; ++r) {
        int s = st * 16 + rowOff + r;
        int d = t * 16 + M;
        atomicAdd(&acc[s * NDP + d], c[st][t][r]);
      }
}

__global__ void dvh_finalize(const float* __restrict__ acc,
                             float* __restrict__ out)
{
  const int lane = threadIdx.x;  // one wave
  float per = 0.0f;
  if (lane < NSM) {
    // tot_vol[s] lives in pad column 85 (GEMM against the all-ones B column).
    float inv = 1.0f / (acc[lane * NDP + NDR] + 1.1920929e-07f);  // FLT_EPSILON
    float ss = 0.0f;
    for (int d = 0; d < NDR; ++d) {
      float x = acc[lane * NDP + d] * inv;
      ss += x * x;
    }
    per = sqrtf(ss);
  }
#pragma unroll
  for (int off = 16; off > 0; off >>= 1) per += __shfl_xor(per, off, 32);
  if (lane == 0) out[0] = per * (1.0f / (85.0f * 22.0f));  // / NSND
}

extern "C" void kernel_launch(void* const* d_in, const int* in_sizes, int n_in,
                              void* d_out, int out_size, void* d_ws, size_t ws_size,
                              hipStream_t stream) {
  const float* y_true = (const float*)d_in[0];
  const float* y_pred = (const float*)d_in[1];
  float* acc = (float*)d_ws;             // [NSP][NDP]
  float* out = (float*)d_out;

  const int N = in_sizes[0] / NSTOT;

  dvh_zero<<<(ACC_ELEMS + 255) / 256, 256, 0, stream>>>(acc);

  const int blocks = 1024;               // 4096 waves, grid-stride over 62.5k chunks
  dvh_accum<<<blocks, BLOCK, 0, stream>>>(y_true, y_pred, acc, N);

  dvh_finalize<<<1, 32, 0, stream>>>(acc, out);
}
